// Seq2SeqAttentionDecoder_80874234184460
// MI455X (gfx1250) — compile-verified, matrix operand order checked by reading, not run on hardware
//
#include <hip/hip_runtime.h>

// ---------------------------------------------------------------------------
// Seq2Seq attention decoder for MI455X (gfx1250), wave32 + WMMA.
// All GEMMs lower to v_wmma_f32_16x16x32_f16 with pure-f16 b128 fragment
// loads (weights converted once per call, activations shadowed in f16).
// ---------------------------------------------------------------------------

typedef __attribute__((ext_vector_type(16))) _Float16 v16h;
typedef __attribute__((ext_vector_type(8)))  _Float16 v8h;
typedef __attribute__((ext_vector_type(8)))  float    v8f;

namespace {

constexpr int kB     = 32;    // batch
constexpr int kT     = 128;   // encoder length
constexpr int kH     = 512;   // hidden
constexpr int kE     = 512;   // embed
constexpr int kOut   = 256;   // output dim
constexpr int kSteps = 16;
constexpr int kG3    = 3 * kH;  // 1536

__device__ __forceinline__ float sigm(float x) { return 1.0f / (1.0f + __expf(-x)); }

// -------------------- fp32 -> f16 conversion / copies ----------------------
__global__ void f32_to_f16_kernel(const float* __restrict__ s,
                                  _Float16* __restrict__ d, int n) {
  int i = blockIdx.x * blockDim.x + threadIdx.x;
  if (i < n) d[i] = (_Float16)s[i];
}

__global__ void copy_f32_kernel(const float* __restrict__ s,
                                float* __restrict__ d, int n) {
  int i = blockIdx.x * blockDim.x + threadIdx.x;
  if (i < n) d[i] = s[i];
}

__global__ void copy_cvt_kernel(const float* __restrict__ s,
                                float* __restrict__ d,
                                _Float16* __restrict__ dh, int n) {
  int i = blockIdx.x * blockDim.x + threadIdx.x;
  if (i < n) { float v = s[i]; d[i] = v; dh[i] = (_Float16)v; }
}

// -------------------- WMMA GEMM: C[M,N] = A[M,K] * W[N,K]^T (+bias) --------
// One wave32 per 16x16 output tile; K stepped in chunks of 32.
// Fragment layout per CDNA5 ISA 7.12.2: per lane the 16 A/B halves are two
// contiguous 16-byte runs -> 4x global_load_b128 per K-chunk, zero converts.
__global__ void gemm_nt_wmma(const _Float16* __restrict__ A,
                             const _Float16* __restrict__ W,
                             const float* __restrict__ bias,
                             float* __restrict__ C,
                             int M, int N, int K) {
  const int gw   = (blockIdx.x * blockDim.x + threadIdx.x) >> 5;
  const int lane = threadIdx.x & 31;
  const int tiles_n = N >> 4;
  const int tiles_m = M >> 4;
  if (gw >= tiles_m * tiles_n) return;   // wave-uniform: EXEC stays all-1s
  const int tm = gw / tiles_n;
  const int tn = gw - tm * tiles_n;

  const int mrow  = (tm << 4) + (lane & 15);   // A row owned by this lane
  const int nrow  = (tn << 4) + (lane & 15);   // W row (== B column) for lane
  const int khalf = (lane & 16) ? 8 : 0;       // lanes 16..31 take K+8 halves

  const _Float16* Arow = A + (size_t)mrow * K + khalf;
  const _Float16* Wrow = W + (size_t)nrow * K + khalf;

  v8f acc = {};
#pragma unroll 2
  for (int k0 = 0; k0 < K; k0 += 32) {
    // WGP-scope prefetch two K-chunks ahead (locality 3 -> near scope);
    // data is L2-resident, so only the near-cache pull is useful.
    __builtin_prefetch(Arow + k0 + 64, 0, 3);   // global_prefetch_b8
    __builtin_prefetch(Wrow + k0 + 64, 0, 3);
    const v8h a0 = *reinterpret_cast<const v8h*>(Arow + k0);
    const v8h a1 = *reinterpret_cast<const v8h*>(Arow + k0 + 16);
    const v8h b0 = *reinterpret_cast<const v8h*>(Wrow + k0);
    const v8h b1 = *reinterpret_cast<const v8h*>(Wrow + k0 + 16);
    const v16h af = __builtin_shufflevector(a0, a1, 0, 1, 2, 3, 4, 5, 6, 7,
                                            8, 9, 10, 11, 12, 13, 14, 15);
    const v16h bf = __builtin_shufflevector(b0, b1, 0, 1, 2, 3, 4, 5, 6, 7,
                                            8, 9, 10, 11, 12, 13, 14, 15);
    acc = __builtin_amdgcn_wmma_f32_16x16x32_f16(
        /*neg_a=*/false, af, /*neg_b=*/false, bf,
        /*c_mod=*/(short)0, acc, /*reuse_a=*/false, /*reuse_b=*/false);
  }

  const int col  = (tn << 4) + (lane & 15);
  const int row0 = (tm << 4) + ((lane >> 4) << 3);  // lanes>=16 hold M=8..15
  const float bv = bias ? bias[col] : 0.0f;
#pragma unroll
  for (int r = 0; r < 8; ++r)
    C[(size_t)(row0 + r) * N + col] = acc[r] + bv;
}

// -------------------- pre[0] = emb(ones) * SOS -----------------------------
__global__ void sos_embed_kernel(const float* __restrict__ embW,
                                 const float* __restrict__ embB,
                                 float* __restrict__ pre0) {
  int i = blockIdx.x * blockDim.x + threadIdx.x;     // b*kE + e
  if (i >= kB * kE) return;
  const int e = i & (kE - 1);
  float s = 0.0f;
  const float* wr = embW + (size_t)e * kOut;
  for (int o = 0; o < kOut; ++o) s += wr[o];
  pre0[i] = (s + embB[e]) * 1.0f;                    // SOS = 1.0
}

// -------------------- attention scores: tanh(q + kproj) . wv ---------------
__global__ void attn_scores_kernel(const float* __restrict__ q,
                                   const float* __restrict__ kproj,
                                   const float* __restrict__ wv,
                                   float* __restrict__ scores) {
  const int gw   = (blockIdx.x * blockDim.x + threadIdx.x) >> 5;  // (b,t)
  const int lane = threadIdx.x & 31;
  if (gw >= kB * kT) return;
  const int b = gw >> 7;
  const float* qp = q + (size_t)b * kH;
  const float* kp = kproj + (size_t)gw * kH;
  float s = 0.0f;
  for (int h = lane; h < kH; h += 32) s += tanhf(qp[h] + kp[h]) * wv[h];
  for (int off = 16; off; off >>= 1) s += __shfl_xor(s, off, 32);
  if (lane == 0) scores[gw] = s;
}

// -------------------- softmax over T=128 (one wave per batch row) ----------
__global__ void attn_softmax_kernel(const float* __restrict__ scores,
                                    float* __restrict__ w,
                                    float* __restrict__ attn_out) {
  const int b = blockIdx.x;
  const int lane = threadIdx.x;                      // blockDim.x == 32
  float v[4];
  float mx = -1e30f;
#pragma unroll
  for (int j = 0; j < 4; ++j) {
    v[j] = scores[b * kT + lane + j * 32];
    mx = fmaxf(mx, v[j]);
  }
  for (int off = 16; off; off >>= 1) mx = fmaxf(mx, __shfl_xor(mx, off, 32));
  float sum = 0.0f;
#pragma unroll
  for (int j = 0; j < 4; ++j) { v[j] = __expf(v[j] - mx); sum += v[j]; }
  for (int off = 16; off; off >>= 1) sum += __shfl_xor(sum, off, 32);
  const float inv = 1.0f / sum;
#pragma unroll
  for (int j = 0; j < 4; ++j) {
    const float val = v[j] * inv;
    w[b * kT + lane + j * 32]        = val;
    attn_out[b * kT + lane + j * 32] = val;
  }
}

// -------------------- context[b,h] = sum_t w[b,t] * enc[b,t,h] -------------
__global__ void attn_context_kernel(const float* __restrict__ w,
                                    const float* __restrict__ enc,
                                    float* __restrict__ ctx) {
  int i = blockIdx.x * blockDim.x + threadIdx.x;     // b*kH + h
  if (i >= kB * kH) return;
  const int b = i >> 9;
  const int h = i & (kH - 1);
  const float* wp = w + (size_t)b * kT;
  const float* ep = enc + (size_t)b * kT * kH + h;
  float s = 0.0f;
  for (int t = 0; t < kT; ++t) s += wp[t] * ep[(size_t)t * kH];
  ctx[i] = s;
}

// -------------------- X[t*32+b, :] = [ctx[b,:] | pre[t,b,:]] (f16) ---------
__global__ void assemble_x_kernel(const float* __restrict__ ctx,
                                  const float* __restrict__ pre,
                                  _Float16* __restrict__ X, int nrows) {
  int i = blockIdx.x * blockDim.x + threadIdx.x;
  if (i >= nrows * (kH + kE)) return;
  const int row = i >> 10;                 // kH+kE == 1024
  const int col = i & 1023;
  const int b = row & (kB - 1);
  float v;
  if (col < kH) v = ctx[b * kH + col];
  else          v = pre[(size_t)row * kE + (col - kH)];
  X[i] = (_Float16)v;
}

// -------------------- GRU gating: h = (1-z)*n + z*h (fp32 + f16 shadow) ----
__global__ void gru_combine_kernel(const float* __restrict__ gi,
                                   const float* __restrict__ gh,
                                   float* __restrict__ h,
                                   _Float16* __restrict__ hh) {
  int i = blockIdx.x * blockDim.x + threadIdx.x;     // b*kH + hx
  if (i >= kB * kH) return;
  const int b = i >> 9;
  const int hx = i & (kH - 1);
  const size_t base = (size_t)b * kG3;
  const float ir = gi[base + hx], iz = gi[base + kH + hx], in = gi[base + 2 * kH + hx];
  const float hr = gh[base + hx], hz = gh[base + kH + hx], hn = gh[base + 2 * kH + hx];
  const float r = sigm(ir + hr);
  const float z = sigm(iz + hz);
  const float n = tanhf(in + r * hn);
  const float o = (1.0f - z) * n + z * h[i];
  h[i]  = o;
  hh[i] = (_Float16)o;
}

// -------------------- teacher forcing select + output write ----------------
__global__ void select_forced_kernel(const float* __restrict__ lin,
                                     const float* __restrict__ target_i,
                                     const int* __restrict__ mask_i,
                                     _Float16* __restrict__ forced16,
                                     float* __restrict__ out_i) {
  int i = blockIdx.x * blockDim.x + threadIdx.x;     // b*kOut + o
  if (i >= kB * kOut) return;
  const int b = i >> 8;
  const float v = (mask_i[b] != 0) ? target_i[i] : lin[i];
  forced16[i] = (_Float16)v;
  out_i[i]    = v;
}

inline int cdiv(int a, int b) { return (a + b - 1) / b; }

inline void launch_gemm(const _Float16* A, const _Float16* W, const float* bias,
                        float* C, int M, int N, int K, hipStream_t s) {
  const int waves = (M >> 4) * (N >> 4);
  const int blocks = cdiv(waves * 32, 256);
  gemm_nt_wmma<<<blocks, 256, 0, s>>>(A, W, bias, C, M, N, K);
}

}  // namespace

extern "C" void kernel_launch(void* const* d_in, const int* in_sizes, int n_in,
                              void* d_out, int out_size, void* d_ws, size_t ws_size,
                              hipStream_t stream) {
  (void)in_sizes; (void)n_in; (void)out_size; (void)ws_size;

  const float* enc    = (const float*)d_in[0];   // [B,T,H]
  const float* hid_in = (const float*)d_in[1];   // [2,B,H]
  const float* target = (const float*)d_in[2];   // [S,B,OUT]
  const float* Wq     = (const float*)d_in[3];
  const float* Wk     = (const float*)d_in[4];
  const float* wv     = (const float*)d_in[5];
  const float* embW   = (const float*)d_in[6];
  const float* embB   = (const float*)d_in[7];
  const float* Wi0    = (const float*)d_in[8];
  const float* Wh0    = (const float*)d_in[9];
  const float* bi0    = (const float*)d_in[10];
  const float* bh0    = (const float*)d_in[11];
  const float* Wi1    = (const float*)d_in[12];
  const float* Wh1    = (const float*)d_in[13];
  const float* bi1    = (const float*)d_in[14];
  const float* bh1    = (const float*)d_in[15];
  const float* lastW  = (const float*)d_in[16];
  const float* lastB  = (const float*)d_in[17];
  const int*   mask   = (const int*)d_in[18];    // [S,B]
  // d_in[19] = num_steps (compile-time kSteps)

  float* out = (float*)d_out;
  float* out_outputs = out;                                     // [S,B,OUT]
  float* out_hidden  = out + kSteps * kB * kOut;                // [2,B,H]
  float* out_attn    = out_hidden + 2 * kB * kH;                // [S,B,1,T]

  // ---------------- workspace carve ----------------
  char* cur = (char*)d_ws;
  auto carve = [&](size_t bytes) -> char* {
    char* p = cur;
    cur += (bytes + 255) & ~(size_t)255;
    return p;
  };
  float* kproj  = (float*)carve(sizeof(float) * kB * kT * kH);
  float* h01    = (float*)carve(sizeof(float) * 2 * kB * kH);   // h0 then h1
  float* qbuf   = (float*)carve(sizeof(float) * kB * kH);
  float* scores = (float*)carve(sizeof(float) * kB * kT);
  float* wbuf   = (float*)carve(sizeof(float) * kB * kT);
  float* ctx    = (float*)carve(sizeof(float) * kB * kH);
  float* pre    = (float*)carve(sizeof(float) * (kSteps + 1) * kB * kE);
  float* Gi0    = (float*)carve(sizeof(float) * (kSteps + 1) * kB * kG3);
  float* Gh     = (float*)carve(sizeof(float) * kB * kG3);
  float* Gi1    = (float*)carve(sizeof(float) * kB * kG3);
  float* Gh1    = (float*)carve(sizeof(float) * kB * kG3);
  float* Lin    = (float*)carve(sizeof(float) * kB * kOut);
  _Float16* h01h   = (_Float16*)carve(sizeof(_Float16) * 2 * kB * kH);
  _Float16* enc16  = (_Float16*)carve(sizeof(_Float16) * kB * kT * kH);
  _Float16* X      = (_Float16*)carve(sizeof(_Float16) * (kSteps + 1) * kB * (kH + kE));
  _Float16* Forced = (_Float16*)carve(sizeof(_Float16) * kB * kOut);
  _Float16* Wi0h   = (_Float16*)carve(sizeof(_Float16) * kG3 * (kE + kH));
  _Float16* Wh0h   = (_Float16*)carve(sizeof(_Float16) * kG3 * kH);
  _Float16* Wi1h   = (_Float16*)carve(sizeof(_Float16) * kG3 * kH);
  _Float16* Wh1h   = (_Float16*)carve(sizeof(_Float16) * kG3 * kH);
  _Float16* Wqh    = (_Float16*)carve(sizeof(_Float16) * kH * kH);
  _Float16* Wkh    = (_Float16*)carve(sizeof(_Float16) * kH * kH);
  _Float16* embWh  = (_Float16*)carve(sizeof(_Float16) * kE * kOut);
  _Float16* lastWh = (_Float16*)carve(sizeof(_Float16) * kOut * kH);

  float*    h0  = h01;
  float*    h1  = h01 + kB * kH;
  _Float16* h0h = h01h;
  _Float16* h1h = h01h + kB * kH;

  // ---------------- one-time (per-call) prep ----------------
  auto cvt = [&](const float* s, _Float16* d, int n) {
    f32_to_f16_kernel<<<cdiv(n, 256), 256, 0, stream>>>(s, d, n);
  };
  cvt(Wi0,  Wi0h,  kG3 * (kE + kH));
  cvt(Wh0,  Wh0h,  kG3 * kH);
  cvt(Wi1,  Wi1h,  kG3 * kH);
  cvt(Wh1,  Wh1h,  kG3 * kH);
  cvt(Wq,   Wqh,   kH * kH);
  cvt(Wk,   Wkh,   kH * kH);
  cvt(embW, embWh, kE * kOut);
  cvt(lastW,lastWh,kOut * kH);
  cvt(enc,  enc16, kB * kT * kH);

  copy_cvt_kernel<<<cdiv(2 * kB * kH, 256), 256, 0, stream>>>(
      hid_in, h01, h01h, 2 * kB * kH);
  sos_embed_kernel<<<cdiv(kB * kE, 256), 256, 0, stream>>>(embW, embB, pre);

  // kproj = enc @ Wk^T  : [B*T, H] x [H, H]^T
  launch_gemm(enc16, Wkh, nullptr, kproj, kB * kT, kH, kH, stream);

  // ---------------- decode steps ----------------
  for (int i = 0; i < kSteps; ++i) {
    // attention
    launch_gemm(h1h, Wqh, nullptr, qbuf, kB, kH, kH, stream);
    attn_scores_kernel<<<cdiv(kB * kT * 32, 256), 256, 0, stream>>>(qbuf, kproj, wv, scores);
    attn_softmax_kernel<<<kB, 32, 0, stream>>>(scores, wbuf, out_attn + (size_t)i * kB * kT);
    attn_context_kernel<<<cdiv(kB * kH, 256), 256, 0, stream>>>(wbuf, enc, ctx);

    // assemble X = [ctx | pre[t]] for whole prefix; batched layer-0 input GEMM
    const int nrows = (i + 1) * kB;
    assemble_x_kernel<<<cdiv(nrows * (kH + kE), 256), 256, 0, stream>>>(ctx, pre, X, nrows);
    launch_gemm(X, Wi0h, bi0, Gi0, nrows, kG3, kE + kH, stream);

    // sequential GRU over the prefix, both layers chained per timestep
    for (int t = 0; t <= i; ++t) {
      launch_gemm(h0h, Wh0h, bh0, Gh, kB, kG3, kH, stream);
      gru_combine_kernel<<<cdiv(kB * kH, 256), 256, 0, stream>>>(
          Gi0 + (size_t)t * kB * kG3, Gh, h0, h0h);
      launch_gemm(h0h, Wi1h, bi1, Gi1, kB, kG3, kH, stream);
      launch_gemm(h1h, Wh1h, bh1, Gh1, kB, kG3, kH, stream);
      gru_combine_kernel<<<cdiv(kB * kH, 256), 256, 0, stream>>>(Gi1, Gh1, h1, h1h);
    }

    // output projection, teacher forcing, next embedding
    launch_gemm(h1h, lastWh, lastB, Lin, kB, kOut, kH, stream);
    select_forced_kernel<<<cdiv(kB * kOut, 256), 256, 0, stream>>>(
        Lin, target + (size_t)i * kB * kOut, mask + i * kB,
        Forced, out_outputs + (size_t)i * kB * kOut);
    launch_gemm(Forced, embWh, embB, pre + (size_t)(i + 1) * kB * kE, kB, kE, kOut, stream);
  }

  // final hidden -> d_out
  copy_f32_kernel<<<cdiv(2 * kB * kH, 256), 256, 0, stream>>>(h01, out_hidden, 2 * kB * kH);
}
